// RLAgent_41858751267165
// MI455X (gfx1250) — compile-verified, matrix operand order checked by reading, not run on hardware
//
#include <hip/hip_runtime.h>
#include <math.h>

// ---------------- CDNA5 WMMA types / helpers ----------------
typedef __attribute__((ext_vector_type(16))) __bf16 v16bf;
typedef __attribute__((ext_vector_type(8)))  float  v8f;

union FragU { v16bf v; unsigned u[8]; uint4 q[2]; };

#define WMMA_BF16(A, Bf, C) \
  __builtin_amdgcn_wmma_f32_16x16x32_bf16(false, (A).v, false, (Bf).v, (short)0, (C), false, false)

__device__ __forceinline__ v8f zero8() {
  v8f z = {0.f, 0.f, 0.f, 0.f, 0.f, 0.f, 0.f, 0.f};
  return z;
}

// round-to-nearest f32 -> bf16 bit pattern
__device__ __forceinline__ unsigned short f2bf(float x) {
  unsigned u = __float_as_uint(x);
  u = (u + 0x7FFFu + ((u >> 16) & 1u)) >> 16;
  return (unsigned short)u;
}

// A-fragment for v_wmma_f32_16x16x32_bf16 from a bf16 row-major (stride 128) buffer.
// lane l holds row m=l&15; element e: K = koff + (e>>3)*16 + (l>>4)*8 + (e&7)
// -> two contiguous 8-element (16B) runs at koff+(l>>4)*8 and +16. 16B-aligned.
__device__ __forceinline__ void load_a_frag(const unsigned short* rowbase, int koff, int lane,
                                            FragU& a) {
  const unsigned short* p = rowbase + koff + ((lane >> 4) << 3);
  a.q[0] = *(const uint4*)(p);
  a.q[1] = *(const uint4*)(p + 16);
}

// Packed-weight B fragment: block of 32x16 bf16 = 1024B, lane reads 32B.
__device__ __forceinline__ void load_b_frag(const uint4* Bp, int block, int lane, FragU& b) {
  const uint4* p = Bp + block * 64 + lane * 2;
  b.q[0] = p[0];
  b.q[1] = p[1];
}

__device__ __forceinline__ float sigf(float x) { return 1.f / (1.f + __expf(-x)); }

// ---------------- problem constants ----------------
#define BATCH   1024
#define N_PTS   50
#define H_DIM   128
#define T_STEPS 50
#define C_TANH_ 10.0f
#define NEG_BIG 1.0e9f

// ---------------- Kernel 0: pack f32 weights into WMMA B-fragment layout ----------------
__global__ void pack_weights(const float* __restrict__ s0, const float* __restrict__ s1,
                             int K, int N, int Ksplit, unsigned short* __restrict__ dst, int total) {
  int idx = blockIdx.x * blockDim.x + threadIdx.x;
  if (idx >= total) return;
  int block = idx >> 9;
  int win = idx & 511;
  int lane = win >> 4;
  int e = win & 15;
  int ktiles = K >> 5;
  int ntile = block / ktiles;
  int ktile = block - ntile * ktiles;
  int k = ktile * 32 + ((e >> 3) << 4) + ((lane >> 4) << 3) + (e & 7);
  int n = ntile * 16 + (lane & 15);
  float v = (k < Ksplit) ? s0[k * N + n] : s1[(k - Ksplit) * N + n];
  dst[idx] = f2bf(v);
}

// ---------------- Kernel 1: context = input_pnt @ W_emb + b_emb, emitted directly as bf16 ----
__global__ void build_context(const float* __restrict__ ip, const float* __restrict__ Wemb,
                              const float* __restrict__ bemb, unsigned short* __restrict__ ctx,
                              int total) {
  int idx = blockIdx.x * blockDim.x + threadIdx.x;
  if (idx >= total) return;
  int bn = idx >> 7;
  int h = idx & 127;
  float v = ip[bn * 2] * Wemb[h] + ip[bn * 2 + 1] * Wemb[H_DIM + h] + bemb[h];
  ctx[idx] = f2bf(v);
}

// ---------------- Kernel 2: C(Mx128,f32) = A(Mx128,bf16) @ Bpacked(128x128,bf16) ----------------
// one wave per 16-row M-tile. Flat k*n loop, double-buffered A and B fragments so the
// scheduler can issue next-iteration loads under the current WMMA (partial loadcnt waits).
__global__ void __launch_bounds__(256) wmma_gemm_bf16(const unsigned short* __restrict__ A,
                                                      const uint4* __restrict__ Bp,
                                                      float* __restrict__ C, int Mtiles) {
  int lane = threadIdx.x & 31;
  int wave = threadIdx.x >> 5;
  int mtile = blockIdx.x * 8 + wave;
  if (mtile >= Mtiles) return;
  const unsigned short* rowbase = A + (size_t)(mtile * 16 + (lane & 15)) * H_DIM;
  v8f acc[8];
#pragma unroll
  for (int n = 0; n < 8; ++n) acc[n] = zero8();

  FragU a[2], b[2];
  load_a_frag(rowbase, 0, lane, a[0]);
  load_b_frag(Bp, 0, lane, b[0]);
#pragma unroll
  for (int i = 0; i < 32; ++i) {          // i = k*8 + n
    int k = i >> 3, n = i & 7;
    if (i < 31) {
      int i2 = i + 1, k2 = i2 >> 3, n2 = i2 & 7;
      if (n2 == 0) load_a_frag(rowbase, k2 * 32, lane, a[k2 & 1]);
      load_b_frag(Bp, n2 * 4 + k2, lane, b[i2 & 1]);
    }
    acc[n] = WMMA_BF16(a[k & 1], b[i & 1], acc[n]);
  }

  int m0 = (lane >> 4) << 3;
  int nc = lane & 15;
#pragma unroll
  for (int n = 0; n < 8; ++n)
#pragma unroll
    for (int r = 0; r < 8; ++r)
      C[(size_t)(mtile * 16 + m0 + r) * H_DIM + n * 16 + nc] = acc[n][r];
}

// per-wave 16x128 @ 128x128 tile GEMM with preloaded B fragments; A fragments all loaded
// up front (LDS, DScnt) so the 4 WMMAs issue back-to-back.
__device__ __forceinline__ v8f gemm_tile_pre(const unsigned short* a_rowbase, const FragU* bw,
                                             int lane) {
  FragU a0, a1, a2, a3;
  load_a_frag(a_rowbase, 0, lane, a0);
  load_a_frag(a_rowbase, 32, lane, a1);
  load_a_frag(a_rowbase, 64, lane, a2);
  load_a_frag(a_rowbase, 96, lane, a3);
  v8f acc = zero8();
  acc = WMMA_BF16(a0, bw[0], acc);
  acc = WMMA_BF16(a1, bw[1], acc);
  acc = WMMA_BF16(a2, bw[2], acc);
  acc = WMMA_BF16(a3, bw[3], acc);
  return acc;
}

// streaming-B variant (off the hot loop: critic only)
__device__ __forceinline__ v8f gemm_tile(const unsigned short* a_rowbase, const uint4* Bp,
                                         int wave, int lane) {
  v8f acc = zero8();
#pragma unroll
  for (int k = 0; k < 4; ++k) {
    FragU a, b;
    load_a_frag(a_rowbase, k * 32, lane, a);
    load_b_frag(Bp, wave * 4 + k, lane, b);
    acc = WMMA_BF16(a, b, acc);
  }
  return acc;
}

// ---------------- Kernel 3: the full sequential decoder (16 batch rows / workgroup) ----------------
__global__ void __launch_bounds__(256)
decoder_kernel(const unsigned short* __restrict__ ctx, const float* __restrict__ e_g,
               const float* __restrict__ e_p,
               const uint4* __restrict__ Wcat, const uint4* __restrict__ Wqg,
               const uint4* __restrict__ Wqp, const uint4* __restrict__ Wrefc,
               const uint4* __restrict__ W1p,
               const float* __restrict__ b_lstm, const float* __restrict__ v_g,
               const float* __restrict__ v_p, const float* __restrict__ dec_init,
               const float* __restrict__ input_pnt, const float* __restrict__ W2,
               const float* __restrict__ b1v, const float* __restrict__ b2v,
               float* __restrict__ outR, float* __restrict__ outV,
               float* __restrict__ outLP, float* __restrict__ outA,
               float* __restrict__ outC, float* __restrict__ outP) {
  __shared__ unsigned short sh_h[16 * H_DIM];    // bf16: WMMA A operand
  __shared__ unsigned short sh_dec[16 * H_DIM];  // bf16: WMMA A operand
  __shared__ unsigned short sh_g[16 * H_DIM];    // bf16: WMMA A operand
  __shared__ float sh_c[16 * H_DIM];             // f32 cell state
  __shared__ float sh_q[16 * H_DIM];             // f32: feeds tanh attention
  __shared__ float sh_u[16 * 64];
  __shared__ float sh_prob[16 * 64];
  __shared__ float sh_mask[16 * 64];
  __shared__ int sh_act[16];

  const int tid = threadIdx.x;
  const int lane = tid & 31;
  const int wave = tid >> 5;
  const int wb = blockIdx.x * 16;  // first batch row of this workgroup

  for (int i = tid; i < 16 * H_DIM; i += 256) {
    sh_c[i] = 0.f;
    sh_h[i] = 0;  // bf16 zero
    sh_dec[i] = f2bf(dec_init[i & (H_DIM - 1)]);
  }
  for (int i = tid; i < 16 * 64; i += 256) sh_mask[i] = 0.f;
  __syncthreads();

  // loop-invariant hoists: per-wave Wq_g / Wq_p B fragments + LSTM biases
  FragU wqg[4], wqp[4];
#pragma unroll
  for (int k = 0; k < 4; ++k) {
    load_b_frag(Wqg, wave * 4 + k, lane, wqg[k]);
    load_b_frag(Wqp, wave * 4 + k, lane, wqp[k]);
  }
  const int m_d = (lane >> 4) << 3;  // D-matrix row base for this lane-half
  const int nc = lane & 15;
  const int hc = 16 * wave + nc;     // hidden column owned by this thread (elementwise)
  const float bi = b_lstm[hc], bfv = b_lstm[H_DIM + hc];
  const float bg = b_lstm[2 * H_DIM + hc], bo = b_lstm[3 * H_DIM + hc];

  const unsigned short* arow_dec = sh_dec + nc * H_DIM;
  const unsigned short* arow_h = sh_h + nc * H_DIM;
  const unsigned short* arow_g = sh_g + nc * H_DIM;

  // tour accumulators live in threads 0..15 (row m = tid)
  float Racc = 0.f, firstx = 0.f, firsty = 0.f, prevx = 0.f, prevy = 0.f;

  for (int t = 0; t < T_STEPS; ++t) {
    // ---- gates = [dec|h](16x256) @ [Wi;Wh](256x512): wave w owns ntiles {w,w+8,w+16,w+24}
    // double-buffered fragments: k+1 loads issue under k's WMMAs.
    v8f accI = zero8(), accF = zero8(), accG = zero8(), accO = zero8();
    {
      FragU a[2], bI[2], bF[2], bG[2], bO[2];
      load_a_frag(arow_dec, 0, lane, a[0]);
      load_b_frag(Wcat, (wave + 0) * 8, lane, bI[0]);
      load_b_frag(Wcat, (wave + 8) * 8, lane, bF[0]);
      load_b_frag(Wcat, (wave + 16) * 8, lane, bG[0]);
      load_b_frag(Wcat, (wave + 24) * 8, lane, bO[0]);
#pragma unroll
      for (int k = 0; k < 8; ++k) {
        int cur = k & 1, nxt = cur ^ 1;
        if (k < 7) {
          int kk = k + 1;
          load_a_frag((kk < 4 ? arow_dec : arow_h), (kk & 3) * 32, lane, a[nxt]);
          load_b_frag(Wcat, (wave + 0) * 8 + kk, lane, bI[nxt]);
          load_b_frag(Wcat, (wave + 8) * 8 + kk, lane, bF[nxt]);
          load_b_frag(Wcat, (wave + 16) * 8 + kk, lane, bG[nxt]);
          load_b_frag(Wcat, (wave + 24) * 8 + kk, lane, bO[nxt]);
        }
        accI = WMMA_BF16(a[cur], bI[cur], accI);
        accF = WMMA_BF16(a[cur], bF[cur], accF);
        accG = WMMA_BF16(a[cur], bG[cur], accG);
        accO = WMMA_BF16(a[cur], bO[cur], accO);
      }
    }
    __syncthreads();  // everyone done reading old h/dec

    // ---- LSTM elementwise: wave w owns hidden cols [16w, 16w+16)
#pragma unroll
    for (int r = 0; r < 8; ++r) {
      int m = m_d + r;
      float gi = accI[r] + bi, gf = accF[r] + bfv, gg = accG[r] + bg, go = accO[r] + bo;
      float co = sh_c[m * H_DIM + hc];
      float cn = sigf(gf) * co + sigf(gi) * tanhf(gg);
      float hn = sigf(go) * tanhf(cn);
      sh_c[m * H_DIM + hc] = cn;
      sh_h[m * H_DIM + hc] = f2bf(hn);
    }
    __syncthreads();

    // ---- q = h2 @ Wq_g -> sh_q (f32, feeds tanh)
    {
      v8f acc = gemm_tile_pre(arow_h, wqg, lane);
#pragma unroll
      for (int r = 0; r < 8; ++r) sh_q[(m_d + r) * H_DIM + hc] = acc[r];
    }
    __syncthreads();

    // ---- glimpse scores u[m,n] = v_g . tanh(e_g + q) - NEG*mask
    //      (prefetch the e_p rows the pointer pass will need: global_prefetch_b8)
    for (int p = tid; p < 16 * N_PTS; p += 256) {
      int m = p / N_PTS, n = p - m * N_PTS;
      const float* er = e_g + ((size_t)(wb + m) * N_PTS + n) * H_DIM;
      const float* pr = e_p + ((size_t)(wb + m) * N_PTS + n) * H_DIM;
      __builtin_prefetch((const void*)pr, 0, 1);
      __builtin_prefetch((const void*)(pr + 64), 0, 1);
      const float* qr = sh_q + m * H_DIM;
      float s = 0.f;
      for (int h = 0; h < H_DIM; ++h) s += tanhf(er[h] + qr[h]) * v_g[h];
      sh_u[m * 64 + n] = s - NEG_BIG * sh_mask[m * 64 + n];
    }
    __syncthreads();

    // ---- softmax over n (row per thread)
    if (tid < 16) {
      int m = tid;
      float mx = -3.4e38f;
      for (int n = 0; n < N_PTS; ++n) mx = fmaxf(mx, sh_u[m * 64 + n]);
      float se = 0.f;
      for (int n = 0; n < N_PTS; ++n) {
        float e = __expf(sh_u[m * 64 + n] - mx);
        sh_prob[m * 64 + n] = e;
        se += e;
      }
      float inv = 1.f / se;
      for (int n = 0; n < N_PTS; ++n) sh_prob[m * 64 + n] *= inv;
    }
    __syncthreads();

    // ---- glimpse vector g = softmax(u) @ e_g  (bf16 out: only feeds WMMA)
    {
      int m = tid >> 4, hb = (tid & 15) * 8;
      float acc[8];
#pragma unroll
      for (int i = 0; i < 8; ++i) acc[i] = 0.f;
      const float* eb = e_g + ((size_t)(wb + m) * N_PTS) * H_DIM + hb;
      for (int n = 0; n < N_PTS; ++n) {
        float pr = sh_prob[m * 64 + n];
        const float* er = eb + n * H_DIM;
#pragma unroll
        for (int i = 0; i < 8; ++i) acc[i] += pr * er[i];
      }
#pragma unroll
      for (int i = 0; i < 8; ++i) sh_g[m * H_DIM + hb + i] = f2bf(acc[i]);
    }
    __syncthreads();

    // ---- gq = g @ Wq_p -> sh_q (f32, feeds tanh)
    {
      v8f acc = gemm_tile_pre(arow_g, wqp, lane);
#pragma unroll
      for (int r = 0; r < 8; ++r) sh_q[(m_d + r) * H_DIM + hc] = acc[r];
    }
    __syncthreads();

    // ---- pointer logits
    for (int p = tid; p < 16 * N_PTS; p += 256) {
      int m = p / N_PTS, n = p - m * N_PTS;
      const float* er = e_p + ((size_t)(wb + m) * N_PTS + n) * H_DIM;
      const float* qr = sh_q + m * H_DIM;
      float s = 0.f;
      for (int h = 0; h < H_DIM; ++h) s += tanhf(er[h] + qr[h]) * v_p[h];
      sh_u[m * 64 + n] = C_TANH_ * tanhf(s) - NEG_BIG * sh_mask[m * 64 + n];
    }
    __syncthreads();

    // ---- log_softmax / softmax / argmax / outputs / mask update
    if (tid < 16) {
      int m = tid, b = wb + m;
      float mx = -3.4e38f;
      for (int n = 0; n < N_PTS; ++n) mx = fmaxf(mx, sh_u[m * 64 + n]);
      float se = 0.f;
      for (int n = 0; n < N_PTS; ++n) se += __expf(sh_u[m * 64 + n] - mx);
      float lse = __logf(se), inv = 1.f / se;
      int amax = 0;
      float bv = sh_u[m * 64];
      for (int n = 1; n < N_PTS; ++n) {
        float x = sh_u[m * 64 + n];
        if (x > bv) { bv = x; amax = n; }
      }
      size_t base = ((size_t)b * T_STEPS + t) * N_PTS;
      for (int n = 0; n < N_PTS; ++n) {
        float x = sh_u[m * 64 + n] - mx;
        outLP[base + n] = x - lse;
        outP[base + n] = __expf(x) * inv;
      }
      outA[(size_t)b * T_STEPS + t] = (float)amax;
      float cx = input_pnt[(b * N_PTS + amax) * 2 + 0];
      float cy = input_pnt[(b * N_PTS + amax) * 2 + 1];
      outC[((size_t)b * T_STEPS + t) * 2 + 0] = cx;
      outC[((size_t)b * T_STEPS + t) * 2 + 1] = cy;
      sh_mask[m * 64 + amax] = 1.f;
      sh_act[m] = amax;
      if (t == 0) {
        firstx = cx; firsty = cy;
      } else {
        float dx = cx - prevx, dy = cy - prevy;
        Racc += sqrtf(dx * dx + dy * dy + 1e-10f);
      }
      prevx = cx; prevy = cy;
    }
    __syncthreads();

    // ---- dec_next = context[b, action] (bf16 -> bf16 straight copy)
    {
      int m = tid >> 4, hb = (tid & 15) * 8;
      const unsigned short* src = ctx + ((size_t)(wb + m) * N_PTS + sh_act[m]) * H_DIM;
#pragma unroll
      for (int i = 0; i < 8; ++i) sh_dec[m * H_DIM + hb + i] = src[hb + i];
    }
    __syncthreads();
  }

  // ---- tour length R (close the loop)
  if (tid < 16) {
    float dx = firstx - prevx, dy = firsty - prevy;
    Racc += sqrtf(dx * dx + dy * dy + 1e-10f);
    outR[wb + tid] = Racc;
  }

  // ---- critic: attention over singleton N collapses exactly to hy = dec_last @ Wref_c
  {
    v8f acc = gemm_tile(arow_dec, Wrefc, wave, lane);
#pragma unroll
    for (int r = 0; r < 8; ++r) sh_h[(m_d + r) * H_DIM + hc] = f2bf(acc[r]);  // hy (bf16)
  }
  __syncthreads();
  {
    v8f acc = gemm_tile(arow_h, W1p, wave, lane);  // hy @ W1
    float bb = b1v[hc];
#pragma unroll
    for (int r = 0; r < 8; ++r) sh_q[(m_d + r) * H_DIM + hc] = fmaxf(acc[r] + bb, 0.f);
  }
  __syncthreads();
  if (tid < 16) {
    float s = 0.f;
    for (int h = 0; h < H_DIM; ++h) s += sh_q[tid * H_DIM + h] * W2[h];
    outV[wb + tid] = s + b2v[0];
  }
}

// ---------------- host-side launch ----------------
extern "C" void kernel_launch(void* const* d_in, const int* in_sizes, int n_in,
                              void* d_out, int out_size, void* d_ws, size_t ws_size,
                              hipStream_t stream) {
  const float* input_pnt = (const float*)d_in[0];
  const float* W_emb = (const float*)d_in[1];
  const float* b_emb = (const float*)d_in[2];
  const float* dec_init = (const float*)d_in[3];
  const float* Wi = (const float*)d_in[4];
  const float* Wh = (const float*)d_in[5];
  const float* b_lstm = (const float*)d_in[6];
  const float* Wq_g = (const float*)d_in[7];
  const float* Wref_g = (const float*)d_in[8];
  const float* v_g = (const float*)d_in[9];
  const float* Wq_p = (const float*)d_in[10];
  const float* Wref_p = (const float*)d_in[11];
  const float* v_p = (const float*)d_in[12];
  // d_in[13..16,18] (Wi_c, Wh_c, b_c, Wq_c, v_c) drop out: softmax over a
  // singleton axis makes the critic attention collapse to dec_last @ Wref_c.
  const float* Wref_c = (const float*)d_in[17];
  const float* W1 = (const float*)d_in[19];
  const float* b1 = (const float*)d_in[20];
  const float* W2 = (const float*)d_in[21];
  const float* b2 = (const float*)d_in[22];

  char* ws = (char*)d_ws;
  unsigned short* ctxb = (unsigned short*)(ws + 0);               // 13,107,200 B (bf16)
  float* e_g = (float*)(ws + 13107200);                           // 26,214,400 B
  float* e_p = (float*)(ws + 39321600);                           // 26,214,400 B
  unsigned short* WcatP = (unsigned short*)(ws + 65536000);       // 262,144 B
  unsigned short* WqgP  = (unsigned short*)(ws + 65798144);       // 32,768 B each below
  unsigned short* WqpP  = (unsigned short*)(ws + 65830912);
  unsigned short* WrefgP = (unsigned short*)(ws + 65863680);
  unsigned short* WrefpP = (unsigned short*)(ws + 65896448);
  unsigned short* WrefcP = (unsigned short*)(ws + 65929216);
  unsigned short* W1P    = (unsigned short*)(ws + 65961984);

  // pack weights into WMMA B-fragment layout (bf16)
  pack_weights<<<512, 256, 0, stream>>>(Wi, Wh, 256, 512, 128, WcatP, 131072);
  pack_weights<<<64, 256, 0, stream>>>(Wq_g, nullptr, 128, 128, 128, WqgP, 16384);
  pack_weights<<<64, 256, 0, stream>>>(Wq_p, nullptr, 128, 128, 128, WqpP, 16384);
  pack_weights<<<64, 256, 0, stream>>>(Wref_g, nullptr, 128, 128, 128, WrefgP, 16384);
  pack_weights<<<64, 256, 0, stream>>>(Wref_p, nullptr, 128, 128, 128, WrefpP, 16384);
  pack_weights<<<64, 256, 0, stream>>>(Wref_c, nullptr, 128, 128, 128, WrefcP, 16384);
  pack_weights<<<64, 256, 0, stream>>>(W1, nullptr, 128, 128, 128, W1P, 16384);

  // context embedding (bf16) + attention key precompute (WMMA GEMMs, f32 out)
  build_context<<<25600, 256, 0, stream>>>(input_pnt, W_emb, b_emb, ctxb, BATCH * N_PTS * H_DIM);
  wmma_gemm_bf16<<<400, 256, 0, stream>>>(ctxb, (const uint4*)WrefgP, e_g, 3200);
  wmma_gemm_bf16<<<400, 256, 0, stream>>>(ctxb, (const uint4*)WrefpP, e_p, 3200);

  // outputs: R(B), v(B), logprobs(B,T,N), actions(B,T), coords(B,T,2), probs(B,T,N)
  float* outR = (float*)d_out;
  float* outV = outR + BATCH;
  float* outLP = outV + BATCH;
  float* outA = outLP + (size_t)BATCH * T_STEPS * N_PTS;
  float* outC = outA + (size_t)BATCH * T_STEPS;
  float* outP = outC + (size_t)BATCH * T_STEPS * 2;

  decoder_kernel<<<64, 256, 0, stream>>>(
      ctxb, e_g, e_p, (const uint4*)WcatP, (const uint4*)WqgP, (const uint4*)WqpP,
      (const uint4*)WrefcP, (const uint4*)W1P, b_lstm, v_g, v_p, dec_init, input_pnt, W2, b1, b2,
      outR, outV, outLP, outA, outC, outP);
}
